// GNNModel_70282844831970
// MI455X (gfx1250) — matrix operand order repaired and checked by reading, not compiled, not run on hardware
//
#include <hip/hip_runtime.h>

#define NODE_IN 128
#define EDGE_IN 64
#define HIDDEN  256

typedef __attribute__((ext_vector_type(16))) __bf16 v16bf;
typedef __attribute__((ext_vector_type(8)))  __bf16 v8bf;
typedef __attribute__((ext_vector_type(8)))  float  v8f;

#define WTILE 512   // one swizzled 32(K)x16(N) weight tile = 32 lanes * 16 halfs

// ---------------------------------------------------------------------------
// Fragment helpers (contiguous k order, same for A and B)
// A 16x32 bf16: lane(m=lane&15, half=lane>>4) holds k = half*16 + s (s = 0..15)
//               of row m  -> two aligned 32B chunks.
// B 32x16 bf16: lane(n=lane&15, khalf=lane>>4) holds k = khalf*16..+15 of col n
//               -> pre-swizzled contiguous 32B per lane in global memory.
// ---------------------------------------------------------------------------
__device__ __forceinline__ v16bf cvt2(v8f lo, v8f hi) {
  v8bf l = __builtin_convertvector(lo, v8bf);
  v8bf h = __builtin_convertvector(hi, v8bf);
  return __builtin_shufflevector(l, h, 0, 1, 2, 3, 4, 5, 6, 7, 8, 9, 10, 11,
                                 12, 13, 14, 15);
}

__device__ __forceinline__ v16bf load_b_swz(const __bf16* __restrict__ wt) {
  v8bf lo = *(const v8bf*)(wt);
  v8bf hi = *(const v8bf*)(wt + 8);
  return __builtin_shufflevector(lo, hi, 0, 1, 2, 3, 4, 5, 6, 7, 8, 9, 10, 11,
                                 12, 13, 14, 15);
}

__device__ __forceinline__ v8f wmma_bf16(v16bf a, v16bf b, v8f c) {
  return __builtin_amdgcn_wmma_f32_16x16x32_bf16(false, a, false, b, (short)0,
                                                 c, false, false);
}

// ---------------------------------------------------------------------------
// Software-pipelined K-loop shared by all WMMA GEMM kernels.
//  - B fragments rotate one stage ahead (loads in flight during WMMAs)
//  - A raw fp32 for kt+1 prefetched at top of kt, converted after the WMMAs
// AP: functor (kt, which) -> const float* row pointer (32 K-floats, 32B align)
// ---------------------------------------------------------------------------
template <int NFRAG, typename AP>
__device__ __forceinline__ void gemm_phase(AP aptr, const __bf16* __restrict__ wl,
                                           int KT, int half,
                                           v8f (&acc)[2][NFRAG]) {
  const float* p0 = aptr(0, 0);
  const float* p1 = aptr(0, 1);
  v8f r00 = *(const v8f*)(p0 + half * 16);
  v8f r01 = *(const v8f*)(p0 + half * 16 + 8);
  v8f r10 = *(const v8f*)(p1 + half * 16);
  v8f r11 = *(const v8f*)(p1 + half * 16 + 8);
  v16bf a0 = cvt2(r00, r01);
  v16bf a1 = cvt2(r10, r11);
  v16bf bcur = load_b_swz(wl);

#pragma unroll 1
  for (int kt = 0; kt < KT; ++kt) {
    const __bf16* wk = wl + (size_t)kt * (NFRAG * 2) * WTILE;
    const bool hasNext = (kt + 1 < KT);
    if (hasNext) {  // prefetch next K-step's A rows (raw fp32)
      const float* q0 = aptr(kt + 1, 0);
      const float* q1 = aptr(kt + 1, 1);
      r00 = *(const v8f*)(q0 + half * 16);
      r01 = *(const v8f*)(q0 + half * 16 + 8);
      r10 = *(const v8f*)(q1 + half * 16);
      r11 = *(const v8f*)(q1 + half * 16 + 8);
    }
#pragma unroll
    for (int f = 0; f < NFRAG; ++f) {
      v16bf bn = bcur;
      if (f + 1 < NFRAG) {
        bn = load_b_swz(wk + (f + 1) * WTILE);
      } else if (hasNext) {
        bn = load_b_swz(wk + (NFRAG * 2) * WTILE);  // (kt+1, f=0)
      }
      acc[0][f] = wmma_bf16(a0, bcur, acc[0][f]);
      acc[1][f] = wmma_bf16(a1, bcur, acc[1][f]);
      bcur = bn;
    }
    if (hasNext) {  // convert prefetched A after the WMMA stream
      a0 = cvt2(r00, r01);
      a1 = cvt2(r10, r11);
    }
  }
}

// ---------------------------------------------------------------------------
// Weight pre-swizzle: W[K][H] fp32 row-major -> bf16 fragment-order tiles.
// out[((kt*(H/16)+nt)*32 + lane)*16 + j] = W[kt*32 + (lane>>4)*16 + j][nt*16 + (lane&15)]
// ---------------------------------------------------------------------------
__global__ __launch_bounds__(256) void swz_kernel(const float* __restrict__ W,
                                                  __bf16* __restrict__ out,
                                                  int K, int H) {
  int t = blockIdx.x * 256 + threadIdx.x;
  if (t >= K * H) return;
  int j = t & 15;
  int lane = (t >> 4) & 31;
  int tile = t >> 9;
  int nt = tile % (H / 16);
  int kt = tile / (H / 16);
  int n = nt * 16 + (lane & 15);
  int k = kt * 32 + ((lane >> 4) << 4) + j;
  out[t] = (__bf16)W[(size_t)k * H + n];
}

// ---------------------------------------------------------------------------
// Scatter (mean aggregation) kernels
// ---------------------------------------------------------------------------
__global__ __launch_bounds__(256) void deg_kernel(const int* __restrict__ dst,
                                                  float* __restrict__ deg, int E) {
  int e = blockIdx.x * 256 + threadIdx.x;
  if (e < E) atomicAdd(&deg[dst[e]], 1.0f);
}

template <int F>
__global__ __launch_bounds__(256) void scatter_kernel(const float* __restrict__ x,
                                                      const int* __restrict__ src,
                                                      const int* __restrict__ dst,
                                                      float* __restrict__ agg, int E) {
  const int per = F / 4;
  int t = blockIdx.x * 256 + threadIdx.x;
  int e = t / per;
  if (e >= E) return;
  int j = (t % per) * 4;
  const float4 v = *(const float4*)(x + (size_t)src[e] * F + j);
  float* a = agg + (size_t)dst[e] * F + j;
  atomicAdd(a + 0, v.x);
  atomicAdd(a + 1, v.y);
  atomicAdd(a + 2, v.z);
  atomicAdd(a + 3, v.w);
}

// ---------------------------------------------------------------------------
// SAGE layer: out = act( (agg*rdeg) @ Wn + x @ Ws + bias )
// Block: 256 thr = 8 waves (4 row-groups x 2 col-groups); wave = 32 rows x
// NFRAG*16 cols; block = 128 rows x NFRAG*32 cols. No LDS, no barriers.
// ---------------------------------------------------------------------------
template <int NFRAG>
__global__ __launch_bounds__(256) void sage_kernel(
    const float* __restrict__ x, const float* __restrict__ agg,
    const float* __restrict__ deg, const __bf16* __restrict__ WnS,
    const __bf16* __restrict__ WsS, const float* __restrict__ bias,
    float* __restrict__ out, int N, int K, int relu) {
  constexpr int H = NFRAG * 32;
  const int tid = threadIdx.x, lane = tid & 31, w = tid >> 5;
  const int rg = w >> 1, cg = w & 1;
  const int half = lane >> 4, m = lane & 15;
  const int rowBase = blockIdx.x * 128;

  int r0 = rowBase + rg * 32 + m;
  int r1 = r0 + 16;
  const int r0c = r0 < N ? r0 : N - 1;
  const int r1c = r1 < N ? r1 : N - 1;

  v8f acc[2][NFRAG];
#pragma unroll
  for (int mi = 0; mi < 2; ++mi)
#pragma unroll
    for (int f = 0; f < NFRAG; ++f)
#pragma unroll
      for (int i = 0; i < 8; ++i) acc[mi][f][i] = 0.0f;

#pragma unroll 1
  for (int s = 0; s < 2; ++s) {
    const float* A = s ? x : agg;
    const __bf16* Wsw = s ? WsS : WnS;
    const __bf16* wl = Wsw + (size_t)(cg * NFRAG) * WTILE + (size_t)lane * 16;
    const float* a0p = A + (size_t)r0c * K;
    const float* a1p = A + (size_t)r1c * K;
    auto ap = [a0p, a1p](int kt, int wch) {
      return (wch ? a1p : a0p) + kt * 32;
    };
    gemm_phase<NFRAG>(ap, wl, K / 32, half, acc);

    if (s == 0) {  // mean = diag(1/max(deg,1)) applied to the agg@Wn partial
#pragma unroll
      for (int mi = 0; mi < 2; ++mi)
#pragma unroll
        for (int i = 0; i < 8; ++i) {
          int row = rowBase + rg * 32 + mi * 16 + half * 8 + i;
          float rd = 1.0f / fmaxf(deg[row < N ? row : N - 1], 1.0f);
#pragma unroll
          for (int f = 0; f < NFRAG; ++f) acc[mi][f][i] *= rd;
        }
    }
  }

#pragma unroll
  for (int mi = 0; mi < 2; ++mi)
#pragma unroll
    for (int f = 0; f < NFRAG; ++f) {
      int col = (cg * NFRAG + f) * 16 + m;
      float bv = bias[col];
#pragma unroll
      for (int i = 0; i < 8; ++i) {
        int row = rowBase + rg * 32 + mi * 16 + half * 8 + i;
        if (row < N) {
          float v = acc[mi][f][i] + bv;
          if (relu) v = fmaxf(v, 0.0f);
          out[(size_t)row * H + col] = v;
        }
      }
    }
}

// ---------------------------------------------------------------------------
// Edge MLP layer 1: h1 = relu( concat(emb[src], emb[dst], ef) @ Mw1 + Mb1 )
// K = 576 = 18 K-steps; region boundaries (256, 512) align with K-steps.
// ---------------------------------------------------------------------------
__global__ __launch_bounds__(256) void edge_mlp1_kernel(
    const float* __restrict__ emb, const int* __restrict__ src,
    const int* __restrict__ dst, const float* __restrict__ ef,
    const __bf16* __restrict__ WS, const float* __restrict__ bias,
    float* __restrict__ out, int eBase, int Ec, int E) {
  constexpr int NFRAG = 8, H = 256;
  constexpr int KT = (2 * HIDDEN + EDGE_IN) / 32;  // 18
  const int tid = threadIdx.x, lane = tid & 31, w = tid >> 5;
  const int rg = w >> 1, cg = w & 1;
  const int half = lane >> 4, m = lane & 15;
  const int rowBase = blockIdx.x * 128;

  int e0 = eBase + rowBase + rg * 32 + m;
  int e1 = e0 + 16;
  if (e0 >= E) e0 = E - 1;
  if (e1 >= E) e1 = E - 1;
  const float* es0 = emb + (size_t)src[e0] * HIDDEN;
  const float* ed0 = emb + (size_t)dst[e0] * HIDDEN;
  const float* ef0 = ef + (size_t)e0 * EDGE_IN;
  const float* es1 = emb + (size_t)src[e1] * HIDDEN;
  const float* ed1 = emb + (size_t)dst[e1] * HIDDEN;
  const float* ef1 = ef + (size_t)e1 * EDGE_IN;

  v8f acc[2][NFRAG];
#pragma unroll
  for (int mi = 0; mi < 2; ++mi)
#pragma unroll
    for (int f = 0; f < NFRAG; ++f)
#pragma unroll
      for (int i = 0; i < 8; ++i) acc[mi][f][i] = 0.0f;

  const __bf16* wl = WS + (size_t)(cg * NFRAG) * WTILE + (size_t)lane * 16;
  auto ap = [&](int kt, int wch) -> const float* {
    int off = kt * 32;
    if (off < HIDDEN) return (wch ? es1 : es0) + off;
    if (off < 2 * HIDDEN) return (wch ? ed1 : ed0) + (off - HIDDEN);
    return (wch ? ef1 : ef0) + (off - 2 * HIDDEN);
  };
  gemm_phase<NFRAG>(ap, wl, KT, half, acc);

#pragma unroll
  for (int mi = 0; mi < 2; ++mi)
#pragma unroll
    for (int f = 0; f < NFRAG; ++f) {
      int col = (cg * NFRAG + f) * 16 + m;
      float bv = bias[col];
#pragma unroll
      for (int i = 0; i < 8; ++i) {
        int r = rowBase + rg * 32 + mi * 16 + half * 8 + i;  // chunk-local
        if (r < Ec) out[(size_t)r * H + col] = fmaxf(acc[mi][f][i] + bv, 0.0f);
      }
    }
}

// ---------------------------------------------------------------------------
// Plain GEMM + bias + optional relu: out[MxH] = act(A[MxK] @ W + b), W swizzled
// ---------------------------------------------------------------------------
template <int NFRAG>
__global__ __launch_bounds__(256) void gemm_kernel(
    const float* __restrict__ Ain, const __bf16* __restrict__ WS,
    const float* __restrict__ bias, float* __restrict__ out, int M, int K,
    int relu) {
  constexpr int H = NFRAG * 32;
  const int tid = threadIdx.x, lane = tid & 31, w = tid >> 5;
  const int rg = w >> 1, cg = w & 1;
  const int half = lane >> 4, m = lane & 15;
  const int rowBase = blockIdx.x * 128;

  int r0 = rowBase + rg * 32 + m;
  int r1 = r0 + 16;
  const int r0c = r0 < M ? r0 : M - 1;
  const int r1c = r1 < M ? r1 : M - 1;
  const float* a0p = Ain + (size_t)r0c * K;
  const float* a1p = Ain + (size_t)r1c * K;

  v8f acc[2][NFRAG];
#pragma unroll
  for (int mi = 0; mi < 2; ++mi)
#pragma unroll
    for (int f = 0; f < NFRAG; ++f)
#pragma unroll
      for (int i = 0; i < 8; ++i) acc[mi][f][i] = 0.0f;

  const __bf16* wl = WS + (size_t)(cg * NFRAG) * WTILE + (size_t)lane * 16;
  auto ap = [a0p, a1p](int kt, int wch) {
    return (wch ? a1p : a0p) + kt * 32;
  };
  gemm_phase<NFRAG>(ap, wl, K / 32, half, acc);

#pragma unroll
  for (int mi = 0; mi < 2; ++mi)
#pragma unroll
    for (int f = 0; f < NFRAG; ++f) {
      int col = (cg * NFRAG + f) * 16 + m;
      float bv = bias[col];
#pragma unroll
      for (int i = 0; i < 8; ++i) {
        int row = rowBase + rg * 32 + mi * 16 + half * 8 + i;
        if (row < M) {
          float v = acc[mi][f][i] + bv;
          if (relu) v = fmaxf(v, 0.0f);
          out[(size_t)row * H + col] = v;
        }
      }
    }
}

// ---------------------------------------------------------------------------
// Final logits: out[e] = h2[e,:] . Mw3 + Mb3   (K = 128)
// ---------------------------------------------------------------------------
__global__ __launch_bounds__(256) void logits_kernel(
    const float* __restrict__ h2, const float* __restrict__ Mw3,
    const float* __restrict__ Mb3, float* __restrict__ out, int Ec) {
  int e = blockIdx.x * 256 + threadIdx.x;
  if (e >= Ec) return;
  const float4* a = (const float4*)(h2 + (size_t)e * (HIDDEN / 2));
  const float4* wv = (const float4*)Mw3;
  float s = 0.0f;
#pragma unroll
  for (int i = 0; i < (HIDDEN / 2) / 4; ++i) {
    float4 u = a[i];
    float4 w = wv[i];
    s += u.x * w.x + u.y * w.y + u.z * w.z + u.w * w.w;
  }
  out[e] = s + Mb3[0];
}

// ---------------------------------------------------------------------------
// Host launcher
// ---------------------------------------------------------------------------
extern "C" void kernel_launch(void* const* d_in, const int* in_sizes, int n_in,
                              void* d_out, int out_size, void* d_ws, size_t ws_size,
                              hipStream_t stream) {
  (void)n_in; (void)out_size; (void)ws_size;
  const float* x   = (const float*)d_in[0];
  const int*   ei  = (const int*)d_in[1];
  const float* ef  = (const float*)d_in[2];
  const float* Wn1 = (const float*)d_in[3];
  const float* Ws1 = (const float*)d_in[4];
  const float* b1  = (const float*)d_in[5];
  const float* Wn2 = (const float*)d_in[6];
  const float* Ws2 = (const float*)d_in[7];
  const float* b2  = (const float*)d_in[8];
  const float* Mw1 = (const float*)d_in[9];
  const float* Mb1 = (const float*)d_in[10];
  const float* Mw2 = (const float*)d_in[11];
  const float* Mb2 = (const float*)d_in[12];
  const float* Mw3 = (const float*)d_in[13];
  const float* Mb3 = (const float*)d_in[14];
  float* outp = (float*)d_out;

  const int N = in_sizes[0] / NODE_IN;   // 50000
  const int E = in_sizes[2] / EDGE_IN;   // 400000
  const int* srcp = ei;
  const int* dstp = ei + E;

  // Workspace carve-out (256B aligned slabs)
  size_t off = 0;
  char* base = (char*)d_ws;
  auto allocB = [&](size_t bytes) -> void* {
    void* p = base + off;
    off += ((bytes + 255) & ~(size_t)255);
    return p;
  };
  float* deg = (float*)allocB((size_t)N * 4);
  float* agg = (float*)allocB((size_t)N * HIDDEN * 4);
  float* h   = (float*)allocB((size_t)N * HIDDEN * 4);
  float* emb = (float*)allocB((size_t)N * HIDDEN * 4);
  const int CH = 64000;  // edge chunk (multiple of 128)
  float* h1c = (float*)allocB((size_t)CH * HIDDEN * 4);
  float* h2c = (float*)allocB((size_t)CH * (HIDDEN / 2) * 4);
  __bf16* wn1s = (__bf16*)allocB((size_t)NODE_IN * HIDDEN * 2);
  __bf16* ws1s = (__bf16*)allocB((size_t)NODE_IN * HIDDEN * 2);
  __bf16* wn2s = (__bf16*)allocB((size_t)HIDDEN * HIDDEN * 2);
  __bf16* ws2s = (__bf16*)allocB((size_t)HIDDEN * HIDDEN * 2);
  __bf16* mw1s = (__bf16*)allocB((size_t)(2 * HIDDEN + EDGE_IN) * HIDDEN * 2);
  __bf16* mw2s = (__bf16*)allocB((size_t)HIDDEN * (HIDDEN / 2) * 2);

  // ---- Pre-swizzle weights to bf16 fragment layout (L2-resident) ----
  auto swz = [&](const float* W, __bf16* o, int K, int H) {
    swz_kernel<<<(K * H + 255) / 256, 256, 0, stream>>>(W, o, K, H);
  };
  swz(Wn1, wn1s, NODE_IN, HIDDEN);
  swz(Ws1, ws1s, NODE_IN, HIDDEN);
  swz(Wn2, wn2s, HIDDEN, HIDDEN);
  swz(Ws2, ws2s, HIDDEN, HIDDEN);
  swz(Mw1, mw1s, 2 * HIDDEN + EDGE_IN, HIDDEN);
  swz(Mw2, mw2s, HIDDEN, HIDDEN / 2);

  // ---- Layer 1: mean aggregation + SAGE GEMM (relu) ----
  hipMemsetAsync(deg, 0, (size_t)N * 4, stream);
  hipMemsetAsync(agg, 0, (size_t)N * HIDDEN * 4, stream);
  deg_kernel<<<(E + 255) / 256, 256, 0, stream>>>(dstp, deg, E);
  scatter_kernel<NODE_IN>
      <<<(unsigned)(((size_t)E * (NODE_IN / 4) + 255) / 256), 256, 0, stream>>>(
          x, srcp, dstp, agg, E);
  sage_kernel<8><<<(N + 127) / 128, 256, 0, stream>>>(
      x, agg, deg, wn1s, ws1s, b1, h, N, NODE_IN, 1);

  // ---- Layer 2: mean aggregation of h + SAGE GEMM (no act) ----
  hipMemsetAsync(agg, 0, (size_t)N * HIDDEN * 4, stream);
  scatter_kernel<HIDDEN>
      <<<(unsigned)(((size_t)E * (HIDDEN / 4) + 255) / 256), 256, 0, stream>>>(
          h, srcp, dstp, agg, E);
  sage_kernel<8><<<(N + 127) / 128, 256, 0, stream>>>(
      h, agg, deg, wn2s, ws2s, b2, emb, N, HIDDEN, 0);

  // ---- Edge classifier, chunked so intermediates stay L2-sized ----
  for (int eBase = 0; eBase < E; eBase += CH) {
    int Ec = (E - eBase < CH) ? (E - eBase) : CH;
    edge_mlp1_kernel<<<(Ec + 127) / 128, 256, 0, stream>>>(
        emb, srcp, dstp, ef, mw1s, Mb1, h1c, eBase, Ec, E);
    gemm_kernel<4><<<(Ec + 127) / 128, 256, 0, stream>>>(
        h1c, mw2s, Mb2, h2c, Ec, HIDDEN, 1);
    logits_kernel<<<(Ec + 255) / 256, 256, 0, stream>>>(
        h2c, Mw3, Mb3, outp + eBase, Ec);
  }
}
